// GATModel_31817117729399
// MI455X (gfx1250) — matrix operand order repaired and checked by reading, not compiled, not run on hardware
//
#include <hip/hip_runtime.h>
#include <hip/hip_bf16.h>

// ---------------------------------------------------------------------------
// Types for CDNA5 WMMA (wave32)
// ---------------------------------------------------------------------------
typedef __attribute__((ext_vector_type(16))) __bf16 v16bf;
typedef __attribute__((ext_vector_type(8)))  float  v8f;

#define NEG_SLOPE 0.2f

// ---------------------------------------------------------------------------
// Helpers
// ---------------------------------------------------------------------------
__device__ __forceinline__ unsigned fenc(float f) {
    // order-preserving encoding of float into unsigned for atomicMax
    unsigned b = __float_as_uint(f);
    return (b & 0x80000000u) ? ~b : (b | 0x80000000u);
}
__device__ __forceinline__ float fdec(unsigned k) {
    return __uint_as_float((k & 0x80000000u) ? (k & 0x7FFFFFFFu) : ~k);
}
__device__ __forceinline__ void edge_sd(const int* __restrict__ ei, int E, long ee,
                                        int& s, int& d) {
    if (ee < (long)E) { s = ei[ee]; d = ei[(long)E + ee]; }
    else              { s = d = (int)(ee - E); }   // self loops appended
}

// ---------------------------------------------------------------------------
// bf16 conversion / weight transpose
// ---------------------------------------------------------------------------
__global__ void cvt_f32_bf16_kernel(const float* __restrict__ in,
                                    __bf16* __restrict__ out, long n) {
    long i = (long)blockIdx.x * blockDim.x + threadIdx.x;
    if (i < n) out[i] = (__bf16)in[i];
}

// W: [K x Nsrc] row-major (f32)  ->  Wt: [Ndst x K] row-major (bf16), zero-padded rows
__global__ void transpose_w_bf16_kernel(const float* __restrict__ W,
                                        __bf16* __restrict__ Wt,
                                        int K, int Nsrc, int Ndst) {
    long i = (long)blockIdx.x * blockDim.x + threadIdx.x;
    if (i >= (long)Ndst * K) return;
    int n = (int)(i / K);
    int k = (int)(i % K);
    Wt[i] = (n < Nsrc) ? (__bf16)W[(size_t)k * Nsrc + n] : (__bf16)0.0f;
}

__global__ void fill_f32_kernel(float* __restrict__ p, float v, long n) {
    long i = (long)blockIdx.x * blockDim.x + threadIdx.x;
    if (i < n) p[i] = v;
}

// ---------------------------------------------------------------------------
// bf16 WMMA GEMM:  C[M x Nc] (f32) = A[M x K] (bf16, row-major)
//                               x Bt[Nc x K] (bf16, row-major = B transposed)
// Block: 256 threads = 8 waves; block tile 128(M) x 64(N); wave tile 16 x 64.
// B tile staged through LDS (reused by all 8 waves).
// ---------------------------------------------------------------------------
__global__ __launch_bounds__(256)
void gemm_bf16_wmma_kernel(const __bf16* __restrict__ A,
                           const __bf16* __restrict__ Bt,
                           float* __restrict__ C,
                           int M, int K, int Nc) {
    const int LDB = 40;                      // bf16 stride: 80B rows, 16B aligned, conflict-free
    __shared__ __bf16 ldsB[64 * 40];         // 64 cols x 32 k, padded

    const int tid  = threadIdx.x;
    const int wid  = tid >> 5;
    const int lane = tid & 31;
    const int bm   = blockIdx.x * 128;
    const int bn   = blockIdx.y * 64;

    // A fragment addressing (16-bit A 16x32 layout; half-lane K split)
    int arow = bm + wid * 16 + (lane & 15);
    if (arow >= M) arow = M - 1;             // clamp; stores predicated later
    const int kA = (lane >> 4) << 3;         // 0 or 8

    // B fragment addressing (lane = column, half-lane K split like A)
    const int kB = (lane >> 4) << 3;         // 0 or 8

    // LDS staging: 256 threads x 8 bf16 = 64x32 tile
    const int sr = tid >> 2;                 // 0..63 (column of B = row of Bt)
    const int sk = (tid & 3) << 3;           // 0,8,16,24
    int brow = bn + sr;
    if (brow >= Nc) brow = Nc - 1;           // clamp (cols >= Nc never stored)

    v8f acc[4];
#pragma unroll
    for (int nb = 0; nb < 4; ++nb)
#pragma unroll
        for (int v = 0; v < 8; ++v) acc[nb][v] = 0.0f;

    for (int k0 = 0; k0 < K; k0 += 32) {
        // ---- stage B tile into LDS (coalesced 16B per thread) ----
        const uint4* gb = (const uint4*)(Bt + (size_t)brow * K + k0 + sk);
        *((uint4*)(ldsB + sr * LDB + sk)) = *gb;
        __syncthreads();

        // ---- A fragment: two 16B loads (K k0+kA..+7 and k0+kA+16..+23) ----
        union { uint4 u[2]; v16bf v; } af;
        const uint4* ga = (const uint4*)(A + (size_t)arow * K + k0 + kA);
        af.u[0] = ga[0];
        af.u[1] = ga[2];

        // ---- 4 B fragments from LDS + 4 WMMAs ----
#pragma unroll
        for (int nb = 0; nb < 4; ++nb) {
            const int coll = nb * 16 + (lane & 15);
            union { uint4 u[2]; v16bf v; } bf;
            const uint4* lb = (const uint4*)(ldsB + coll * LDB + kB);
            bf.u[0] = lb[0];
            bf.u[1] = lb[2];
            acc[nb] = __builtin_amdgcn_wmma_f32_16x16x32_bf16(
                false, af.v, false, bf.v, (short)0, acc[nb], false, false);
        }
        __syncthreads();
    }

    // ---- store D: VGPR v -> row v (+8 for upper half-lanes), col = lane&15 ----
    const int crow0 = bm + wid * 16 + ((lane >> 4) << 3);
    const int ccol  = lane & 15;
#pragma unroll
    for (int nb = 0; nb < 4; ++nb) {
        const int col = bn + nb * 16 + ccol;
        if (col < Nc) {
#pragma unroll
            for (int v = 0; v < 8; ++v) {
                const int rr = crow0 + v;
                if (rr < M) C[(size_t)rr * Nc + col] = acc[nb][v];
            }
        }
    }
}

// ---------------------------------------------------------------------------
// Per-node attention contributions: als/ald [N x H]
// ---------------------------------------------------------------------------
__global__ void attn_node_kernel(const float* __restrict__ Hf,
                                 const float* __restrict__ a_s,
                                 const float* __restrict__ a_d,
                                 float* __restrict__ als,
                                 float* __restrict__ ald,
                                 int N, int H, int hid, int stride) {
    long idx = (long)blockIdx.x * blockDim.x + threadIdx.x;
    if (idx >= (long)N * H) return;
    const int n = (int)(idx / H);
    const int h = (int)(idx % H);
    const float* row = Hf + (size_t)n * stride + h * hid;
    float s = 0.0f, d = 0.0f;
    for (int c = 0; c < hid; ++c) {
        const float v = row[c];
        s += v * a_s[h * hid + c];
        d += v * a_d[h * hid + c];
    }
    als[idx] = s;
    ald[idx] = d;
}

// ---------------------------------------------------------------------------
// Edge pass 1: logits + LeakyReLU, segment-max via encoded atomicMax
// ---------------------------------------------------------------------------
__global__ void edge_logits_kernel(const int* __restrict__ ei, int E, int H,
                                   const float* __restrict__ als,
                                   const float* __restrict__ ald,
                                   float* __restrict__ eBuf,
                                   unsigned* __restrict__ mEnc, long total) {
    long idx = (long)blockIdx.x * blockDim.x + threadIdx.x;
    if (idx >= total) return;
    const long ee = idx / H;
    const int  h  = (int)(idx % H);
    int s, d; edge_sd(ei, E, ee, s, d);
    float e = als[(size_t)s * H + h] + ald[(size_t)d * H + h];
    e = (e > 0.0f) ? e : NEG_SLOPE * e;
    eBuf[idx] = e;
    atomicMax(&mEnc[(size_t)d * H + h], fenc(e));
}

// ---------------------------------------------------------------------------
// Edge pass 2: ex = exp(e - max), denom accumulation
// ---------------------------------------------------------------------------
__global__ void edge_exp_kernel(const int* __restrict__ ei, int E, int H,
                                float* __restrict__ eBuf,
                                const unsigned* __restrict__ mEnc,
                                float* __restrict__ denom, long total) {
    long idx = (long)blockIdx.x * blockDim.x + threadIdx.x;
    if (idx >= total) return;
    const long ee = idx / H;
    const int  h  = (int)(idx % H);
    int s, d; edge_sd(ei, E, ee, s, d);
    const float mm = fdec(mEnc[(size_t)d * H + h]);
    const float ex = __expf(eBuf[idx] - mm);
    eBuf[idx] = ex;
    atomicAdd(&denom[(size_t)d * H + h], ex);
}

// ---------------------------------------------------------------------------
// Edge pass 3: alpha-weighted scatter-aggregation
// ---------------------------------------------------------------------------
__global__ void aggregate_kernel(const int* __restrict__ ei, int E, int H,
                                 int hid, int Cv, int stride,
                                 const float* __restrict__ Hf,
                                 const float* __restrict__ eBuf,
                                 const float* __restrict__ denom,
                                 float* __restrict__ agg, long total) {
    long idx = (long)blockIdx.x * blockDim.x + threadIdx.x;
    if (idx >= total) return;
    const long ee = idx / Cv;
    const int  j  = (int)(idx % Cv);
    const int  h  = j / hid;
    int s, d; edge_sd(ei, E, ee, s, d);
    const float alpha = eBuf[ee * H + h] /
                        (denom[(size_t)d * H + h] + 1e-16f);
    atomicAdd(&agg[(size_t)d * stride + j], alpha * Hf[(size_t)s * stride + j]);
}

// ---------------------------------------------------------------------------
// Fused bias + ELU + f32->bf16 (next layer GEMM input)
// ---------------------------------------------------------------------------
__global__ void bias_elu_bf16_kernel(const float* __restrict__ agg,
                                     const float* __restrict__ b,
                                     __bf16* __restrict__ out, long n, int C) {
    long i = (long)blockIdx.x * blockDim.x + threadIdx.x;
    if (i >= n) return;
    const int j = (int)(i % C);
    float v = agg[i] + b[j];
    v = (v > 0.0f) ? v : expm1f(v);
    out[i] = (__bf16)v;
}

// ---------------------------------------------------------------------------
// Mean pooling of final layer (+bias) into pool[NCLS]
// ---------------------------------------------------------------------------
__global__ void pool_kernel(const float* __restrict__ agg,
                            const float* __restrict__ b,
                            float* __restrict__ pool,
                            int N, int stride, int Cv, float invN) {
    long i = (long)blockIdx.x * blockDim.x + threadIdx.x;
    if (i >= (long)N * Cv) return;
    const int n = (int)(i / Cv);
    const int c = (int)(i % Cv);
    atomicAdd(&pool[c], (agg[(size_t)n * stride + c] + b[c]) * invN);
}

__global__ void log_softmax_kernel(const float* __restrict__ pool,
                                   float* __restrict__ out, int C) {
    if (blockIdx.x == 0 && threadIdx.x == 0) {
        float mx = -1e30f;
        for (int c = 0; c < C; ++c) mx = fmaxf(mx, pool[c]);
        float s = 0.0f;
        for (int c = 0; c < C; ++c) s += __expf(pool[c] - mx);
        const float ls = logf(s);
        for (int c = 0; c < C; ++c) out[c] = pool[c] - mx - ls;
    }
}

// ---------------------------------------------------------------------------
// Host side
// ---------------------------------------------------------------------------
static inline void* wsa(char*& p, size_t bytes) {
    void* r = (void*)p;
    p += (bytes + 255) & ~(size_t)255;
    return r;
}
static inline unsigned gb(long n, int bs) { return (unsigned)((n + bs - 1) / bs); }

extern "C" void kernel_launch(void* const* d_in, const int* in_sizes, int n_in,
                              void* d_out, int out_size, void* d_ws, size_t ws_size,
                              hipStream_t stream) {
    const float* x      = (const float*)d_in[0];
    const int*   ei     = (const int*)  d_in[1];
    const float* W1     = (const float*)d_in[2];
    const float* a_s1   = (const float*)d_in[3];
    const float* a_d1   = (const float*)d_in[4];
    const float* b1     = (const float*)d_in[5];
    const float* W2     = (const float*)d_in[6];
    const float* a_s2   = (const float*)d_in[7];
    const float* a_d2   = (const float*)d_in[8];
    const float* b2     = (const float*)d_in[9];
    const float* W3     = (const float*)d_in[10];
    const float* a_s3   = (const float*)d_in[11];
    const float* a_d3   = (const float*)d_in[12];
    const float* b3     = (const float*)d_in[13];

    const int F_IN = 128, HEADS = 4, HID = 64, C = 256, NCLS = 10, NPAD = 16;
    const int N    = in_sizes[0] / F_IN;
    const int E    = in_sizes[1] / 2;
    const int Etot = E + N;                   // with self loops
    float* outp    = (float*)d_out;

    // workspace carve-up
    char* p = (char*)d_ws;
    __bf16*   xb    = (__bf16*)  wsa(p, (size_t)N * F_IN * sizeof(__bf16));
    __bf16*   Wt1   = (__bf16*)  wsa(p, (size_t)C * F_IN * sizeof(__bf16));
    __bf16*   Wt2   = (__bf16*)  wsa(p, (size_t)C * C * sizeof(__bf16));
    __bf16*   Wt3   = (__bf16*)  wsa(p, (size_t)NPAD * C * sizeof(__bf16));
    float*    Hf    = (float*)   wsa(p, (size_t)N * C * sizeof(float));
    float*    agg   = (float*)   wsa(p, (size_t)N * C * sizeof(float));
    __bf16*   hb    = (__bf16*)  wsa(p, (size_t)N * C * sizeof(__bf16));
    float*    als   = (float*)   wsa(p, (size_t)N * HEADS * sizeof(float));
    float*    ald   = (float*)   wsa(p, (size_t)N * HEADS * sizeof(float));
    unsigned* mEnc  = (unsigned*)wsa(p, (size_t)N * HEADS * sizeof(unsigned));
    float*    denom = (float*)   wsa(p, (size_t)N * HEADS * sizeof(float));
    float*    eBuf  = (float*)   wsa(p, (size_t)Etot * HEADS * sizeof(float));
    float*    pool  = (float*)   wsa(p, 64);

    const int BS = 256;

    // --- prep: conversions + weight transposes ---
    cvt_f32_bf16_kernel<<<gb((long)N * F_IN, BS), BS, 0, stream>>>(x, xb, (long)N * F_IN);
    transpose_w_bf16_kernel<<<gb((long)C * F_IN, BS), BS, 0, stream>>>(W1, Wt1, F_IN, C, C);
    transpose_w_bf16_kernel<<<gb((long)C * C, BS), BS, 0, stream>>>(W2, Wt2, C, C, C);
    transpose_w_bf16_kernel<<<gb((long)NPAD * C, BS), BS, 0, stream>>>(W3, Wt3, C, NCLS, NPAD);

    // ======================== layer 1 ========================
    {
        dim3 grid(gb(N, 128), C / 64);
        gemm_bf16_wmma_kernel<<<grid, 256, 0, stream>>>(xb, Wt1, Hf, N, F_IN, C);
        attn_node_kernel<<<gb((long)N * HEADS, BS), BS, 0, stream>>>(
            Hf, a_s1, a_d1, als, ald, N, HEADS, HID, C);
        fill_f32_kernel<<<gb((long)N * HEADS, BS), BS, 0, stream>>>((float*)mEnc, 0.0f, (long)N * HEADS);
        fill_f32_kernel<<<gb((long)N * HEADS, BS), BS, 0, stream>>>(denom, 0.0f, (long)N * HEADS);
        const long EH = (long)Etot * HEADS;
        edge_logits_kernel<<<gb(EH, BS), BS, 0, stream>>>(ei, E, HEADS, als, ald, eBuf, mEnc, EH);
        edge_exp_kernel<<<gb(EH, BS), BS, 0, stream>>>(ei, E, HEADS, eBuf, mEnc, denom, EH);
        fill_f32_kernel<<<gb((long)N * C, BS), BS, 0, stream>>>(agg, 0.0f, (long)N * C);
        const long EC = (long)Etot * C;
        aggregate_kernel<<<gb(EC, BS), BS, 0, stream>>>(ei, E, HEADS, HID, C, C, Hf, eBuf, denom, agg, EC);
        bias_elu_bf16_kernel<<<gb((long)N * C, BS), BS, 0, stream>>>(agg, b1, hb, (long)N * C, C);
    }

    // ======================== layer 2 ========================
    {
        dim3 grid(gb(N, 128), C / 64);
        gemm_bf16_wmma_kernel<<<grid, 256, 0, stream>>>(hb, Wt2, Hf, N, C, C);
        attn_node_kernel<<<gb((long)N * HEADS, BS), BS, 0, stream>>>(
            Hf, a_s2, a_d2, als, ald, N, HEADS, HID, C);
        fill_f32_kernel<<<gb((long)N * HEADS, BS), BS, 0, stream>>>((float*)mEnc, 0.0f, (long)N * HEADS);
        fill_f32_kernel<<<gb((long)N * HEADS, BS), BS, 0, stream>>>(denom, 0.0f, (long)N * HEADS);
        const long EH = (long)Etot * HEADS;
        edge_logits_kernel<<<gb(EH, BS), BS, 0, stream>>>(ei, E, HEADS, als, ald, eBuf, mEnc, EH);
        edge_exp_kernel<<<gb(EH, BS), BS, 0, stream>>>(ei, E, HEADS, eBuf, mEnc, denom, EH);
        fill_f32_kernel<<<gb((long)N * C, BS), BS, 0, stream>>>(agg, 0.0f, (long)N * C);
        const long EC = (long)Etot * C;
        aggregate_kernel<<<gb(EC, BS), BS, 0, stream>>>(ei, E, HEADS, HID, C, C, Hf, eBuf, denom, agg, EC);
        bias_elu_bf16_kernel<<<gb((long)N * C, BS), BS, 0, stream>>>(agg, b2, hb, (long)N * C, C);
    }

    // ======================== layer 3 (heads=1, 10 classes padded to 16) ====
    {
        dim3 grid(gb(N, 128), 1);
        gemm_bf16_wmma_kernel<<<grid, 256, 0, stream>>>(hb, Wt3, Hf, N, C, NPAD);
        attn_node_kernel<<<gb((long)N, BS), BS, 0, stream>>>(
            Hf, a_s3, a_d3, als, ald, N, 1, NCLS, NPAD);
        fill_f32_kernel<<<gb((long)N, BS), BS, 0, stream>>>((float*)mEnc, 0.0f, (long)N);
        fill_f32_kernel<<<gb((long)N, BS), BS, 0, stream>>>(denom, 0.0f, (long)N);
        const long EH = (long)Etot;
        edge_logits_kernel<<<gb(EH, BS), BS, 0, stream>>>(ei, E, 1, als, ald, eBuf, mEnc, EH);
        edge_exp_kernel<<<gb(EH, BS), BS, 0, stream>>>(ei, E, 1, eBuf, mEnc, denom, EH);
        fill_f32_kernel<<<gb((long)N * NPAD, BS), BS, 0, stream>>>(agg, 0.0f, (long)N * NPAD);
        const long EC = (long)Etot * NCLS;
        aggregate_kernel<<<gb(EC, BS), BS, 0, stream>>>(ei, E, 1, NCLS, NCLS, NPAD, Hf, eBuf, denom, agg, EC);
        fill_f32_kernel<<<1, 64, 0, stream>>>(pool, 0.0f, NCLS);
        pool_kernel<<<gb((long)N * NCLS, BS), BS, 0, stream>>>(agg, b3, pool, N, NPAD, NCLS, 1.0f / (float)N);
        log_softmax_kernel<<<1, 32, 0, stream>>>(pool, outp, NCLS);
    }
}
